// RobertaLayer_22454089024067
// MI455X (gfx1250) — matrix operand order repaired and checked by reading, not compiled, not run on hardware
//
#include <hip/hip_runtime.h>
#include <hip/hip_fp16.h>
#include <math.h>

// ---------------------------------------------------------------------------
// RoBERTa layer forward for MI455X (gfx1250):
//   - f16 WMMA GEMMs (v_wmma_f32_16x16x32_f16)
//   - TDM (tensor_load_to_lds) stages GEMM weight tiles, with TDM LDS padding
//     producing the bank-conflict-free 80B pitch directly
//   - global_load_async_to_lds_b128 stages GEMM A tiles and attention K/V
//   - fused flash attention (online softmax, no materialized score matrix)
// ---------------------------------------------------------------------------

typedef __attribute__((ext_vector_type(16))) _Float16 v16h;
typedef __attribute__((ext_vector_type(8)))  float    v8f;
typedef unsigned int u32x4 __attribute__((ext_vector_type(4)));
typedef int          i32x8 __attribute__((ext_vector_type(8)));
typedef int          i32x4 __attribute__((ext_vector_type(4)));

constexpr int H  = 1024;
constexpr int NH = 16;
constexpr int HD = 64;
constexpr int FF = 4096;
constexpr int Bb = 2;
constexpr int S  = 2048;
constexpr int MT = Bb * S;            // 4096 tokens
constexpr float LN_EPS = 1e-12f;

union FragH { v16h v; uint4 q[2]; };

__device__ __forceinline__ v8f wmma16x16x32(v16h a, v16h b, v8f c) {
  return __builtin_amdgcn_wmma_f32_16x16x32_f16(false, a, false, b,
                                                (short)0, c, false, false);
}

// Raw LDS byte offset of a __shared__ pointer (flat LDS addr low 32 bits).
__device__ __forceinline__ uint32_t lds_off(const void* p) {
  return (uint32_t)(uintptr_t)p;
}

// Per-lane async DMA: 16B global -> LDS, tracked by ASYNCcnt.
__device__ __forceinline__ void async_ld_b128(uint32_t lds, const void* g) {
  asm volatile("global_load_async_to_lds_b128 %0, %1, off"
               :: "v"(lds), "v"((unsigned long long)(uintptr_t)g)
               : "memory");
}
__device__ __forceinline__ void wait_async0() {
  asm volatile("s_wait_asynccnt 0x0" ::: "memory");
}
__device__ __forceinline__ void wait_tensor0() {
  __builtin_amdgcn_s_wait_tensorcnt(0);
}

// ---------------------------------------------------------------------------
// TDM: load a 2D f16 tile (rows x 32 halfs, row stride = rowstride elements)
// from global to LDS.  TDM pad: after every 64B row insert 16B -> 80B pitch.
// D# built per CDNA5 ISA 8.3/8.4 (group0: flags/lds/global, group1: dims).
// ---------------------------------------------------------------------------
__device__ __forceinline__ void tdm_load_tile_f16(const _Float16* g,
                                                  uint32_t lds,
                                                  int rows, int rowstride)
{
  const unsigned long long ga = (unsigned long long)(uintptr_t)g;
  u32x4 g0;
  g0[0] = 1u;                                            // count=1, user desc
  g0[1] = lds;                                           // lds_addr
  g0[2] = (uint32_t)ga;                                  // global_addr[31:0]
  g0[3] = (uint32_t)((ga >> 32) & 0x01FFFFFFu)           // global_addr[56:32]
        | (2u << 30);                                    // type=2 (image)
  const uint32_t rowlen = 32;                            // tile_dim0 (halfs)
  i32x8 g1;
  g1[0] = (int)((1u << 16)     // data_size = 2B
              | (1u << 20)     // pad_enable
              | (3u << 22)     // pad_interval: 16 DWORDs (64B)
              | (3u << 25));   // pad_amount:   4 DWORDs (16B)
  g1[1] = (int)(rowlen << 16);                           // tensor_dim0[15:0]
  g1[2] = (int)((uint32_t)rows << 16);                   // tensor_dim1[15:0]
  g1[3] = (int)(rowlen << 16);                           // tile_dim0
  g1[4] = (int)(uint32_t)rows;                           // tile_dim1 (dim2=0)
  g1[5] = (int)(uint32_t)rowstride;                      // dim0_stride[31:0]
  g1[6] = 0;
  g1[7] = 0;
  const i32x4 gz = {0, 0, 0, 0};                         // dims 2-4 unused
  asm volatile("tensor_load_to_lds %0, %1, %2, %3"
               :: "s"(g0), "s"(g1), "s"(gz), "s"(gz)
               : "memory");
}

// ---------------------------------------------------------------------------
// Tiled WMMA GEMM:  C[M,N] = epi(A[M,K] * Wt[N,K]^T + bias[N])
//   A tile  (128x32): per-lane async global->LDS
//   Wt tile (128x32): one TDM descriptor per k-step (wave 0 issues)
// ---------------------------------------------------------------------------
constexpr int TM = 128, TN = 128, TK = 32;
constexpr int LDT = TK + 8;          // 40 halfs = 80B pitch (TDM-padded too)

enum { EPI_QK = 0, EPI_VT = 1, EPI_RES = 2, EPI_GELU = 3 };

template <int EPI>
__global__ __launch_bounds__(256)
void gemm_wmma(const _Float16* __restrict__ A,    // [Mdim,Kdim] f16
               const _Float16* __restrict__ Wt,   // [Ndim,Kdim] f16
               const float*    __restrict__ bias, // [Ndim]
               const float*    __restrict__ res,  // [Mdim,Ndim] or null
               float*          __restrict__ outF,
               _Float16*       __restrict__ outH,
               int Mdim, int Ndim, int Kdim)
{
  __shared__ _Float16 sA[2][TM * LDT];
  __shared__ _Float16 sB[2][TN * LDT];

  const int tid = threadIdx.x;
  const int m0  = blockIdx.y * TM;
  const int n0  = blockIdx.x * TN;

  // A staging: 2 threads/row, 16 halfs (32B) per thread -> 2 async b128 each
  const int srow = tid >> 1;
  const int sseg = (tid & 1) * 16;
  const _Float16* gA  = A  + (size_t)(m0 + srow) * Kdim + sseg;
  const _Float16* gBt = Wt + (size_t)n0 * Kdim;
  uint32_t ldsA[2] = { lds_off(&sA[0][srow * LDT + sseg]),
                       lds_off(&sA[1][srow * LDT + sseg]) };
  uint32_t ldsB[2] = { lds_off(&sB[0][0]), lds_off(&sB[1][0]) };

  const int wave = tid >> 5;
  const int wm   = wave >> 1;         // 0..3 : 32-row slice
  const int wn   = wave & 1;          // 0..1 : 64-col slice
  const int lane = tid & 31;
  const int r    = lane & 15;
  const int half = lane >> 4;

  auto stage = [&](int ks, int b) {
    const _Float16* ga = gA + (size_t)ks * TK;
    async_ld_b128(ldsA[b],      ga);
    async_ld_b128(ldsA[b] + 16, ga + 8);
    if (wave == 0)
      tdm_load_tile_f16(gBt + (size_t)ks * TK, ldsB[b], TN, Kdim);
  };

  const v8f vz = {0.f, 0.f, 0.f, 0.f, 0.f, 0.f, 0.f, 0.f};
  v8f acc[2][4];
  for (int i = 0; i < 2; ++i)
    for (int j = 0; j < 4; ++j) acc[i][j] = vz;

  const int nk = Kdim / TK;
  stage(0, 0);
  wait_async0();
  wait_tensor0();
  __syncthreads();

#pragma unroll 1
  for (int ks = 0; ks < nk; ++ks) {
    const int buf = ks & 1;
    if (ks + 1 < nk) stage(ks + 1, buf ^ 1);   // DMA overlaps the math below

    FragH af[2], bf[4];
#pragma unroll
    for (int i = 0; i < 2; ++i) {
      const _Float16* p = &sA[buf][(wm * 32 + 16 * i + r) * LDT];
      af[i].q[0] = *(const uint4*)(p + half * 8);
      af[i].q[1] = *(const uint4*)(p + 16 + half * 8);
    }
#pragma unroll
    for (int j = 0; j < 4; ++j) {
      const _Float16* p = &sB[buf][(wn * 64 + 16 * j + r) * LDT];
      bf[j].q[0] = *(const uint4*)(p + half * 8);
      bf[j].q[1] = *(const uint4*)(p + 16 + half * 8);
    }
#pragma unroll
    for (int i = 0; i < 2; ++i)
#pragma unroll
      for (int j = 0; j < 4; ++j)
        acc[i][j] = wmma16x16x32(af[i].v, bf[j].v, acc[i][j]);

    wait_async0();
    wait_tensor0();
    __syncthreads();
  }

  // C layout: VGPR v -> row v + 8*half, col r
#pragma unroll
  for (int i = 0; i < 2; ++i) {
#pragma unroll
    for (int j = 0; j < 4; ++j) {
#pragma unroll
      for (int v = 0; v < 8; ++v) {
        const int m = m0 + wm * 32 + 16 * i + v + 8 * half;
        const int n = n0 + wn * 64 + 16 * j + r;
        const float val = acc[i][j][v] + bias[n];
        if (EPI == EPI_QK || EPI == EPI_VT) {
          const int hh = n >> 6;
          const int dd = n & 63;
          const int bi2 = m / S;
          const int si2 = m - bi2 * S;
          size_t idx;
          if (EPI == EPI_QK)
            idx = ((size_t)(bi2 * NH + hh) * S + si2) * HD + dd;
          else
            idx = ((size_t)(bi2 * NH + hh) * HD + dd) * S + si2;
          outH[idx] = (_Float16)val;
        } else if (EPI == EPI_RES) {
          const size_t idx = (size_t)m * Ndim + n;
          outF[idx] = val + res[idx];
        } else {  // EPI_GELU, exact erf form
          const float g = 0.5f * val * (1.0f + erff(val * 0.70710678118654752f));
          outH[(size_t)m * Ndim + n] = (_Float16)g;
        }
      }
    }
  }
}

// ---------------------------------------------------------------------------
// Fused flash attention.  Per (b,h): 128 queries per WG, 8 waves x 16 rows.
// K/V blocks (32 keys) double-buffered in LDS via async-to-LDS.
// qh/kh: [B,NH,S,HD] f16, vt: [B,NH,HD,S] f16, out ctxh: [B*S, H] f16.
// ---------------------------------------------------------------------------
constexpr int KP = 72;   // sK pitch (halfs): 144B = 9x16B, conflict-free
constexpr int VP = 40;   // sV/sP pitch (halfs): 80B

__global__ __launch_bounds__(256)
void attn_flash(const _Float16* __restrict__ qh,
                const _Float16* __restrict__ kh,
                const _Float16* __restrict__ vt,
                const int*      __restrict__ amask,  // [B,S]
                _Float16*       __restrict__ ctxh)
{
  __shared__ _Float16 sK[2][32 * KP];
  __shared__ _Float16 sV[2][64 * VP];
  __shared__ _Float16 sP[8][16 * VP];

  const int tid  = threadIdx.x;
  const int wave = tid >> 5;
  const int lane = tid & 31;
  const int r    = lane & 15;
  const int half = lane >> 4;

  const int bh = blockIdx.y;
  const int bb = bh / NH;
  const int hh = bh - bb * NH;
  const int q0 = blockIdx.x * 128 + wave * 16;

  const _Float16* Q = qh + (size_t)bh * S * HD;
  const _Float16* K = kh + (size_t)bh * S * HD;
  const _Float16* V = vt + (size_t)bh * HD * S;
  const int* mk = amask + bb * S;

  // staging map: K tile 32x64 halfs (8 chunks/row), V tile 64x32 (4 chunks/row)
  const int krow = tid >> 3, kseg = (tid & 7) * 8;
  const int vrow = tid >> 2, vseg = (tid & 3) * 8;
  uint32_t ldsK[2] = { lds_off(&sK[0][krow * KP + kseg]),
                       lds_off(&sK[1][krow * KP + kseg]) };
  uint32_t ldsV[2] = { lds_off(&sV[0][vrow * VP + vseg]),
                       lds_off(&sV[1][vrow * VP + vseg]) };
  const _Float16* gK = K + (size_t)krow * HD + kseg;
  const _Float16* gV = V + (size_t)vrow * S + vseg;

  auto stageKV = [&](int kt, int b) {
    async_ld_b128(ldsK[b], gK + (size_t)kt * HD);
    async_ld_b128(ldsV[b], gV + kt);
  };

  // Q fragments for this wave's 16 rows (two 32-wide k-steps over HD=64)
  FragH qf[2];
#pragma unroll
  for (int s2 = 0; s2 < 2; ++s2) {
    const _Float16* p = Q + (size_t)(q0 + r) * HD + 32 * s2;
    qf[s2].q[0] = *(const uint4*)(p + half * 8);
    qf[s2].q[1] = *(const uint4*)(p + 16 + half * 8);
  }

  const v8f vz = {0.f, 0.f, 0.f, 0.f, 0.f, 0.f, 0.f, 0.f};
  v8f ctx[4];
#pragma unroll
  for (int t = 0; t < 4; ++t) ctx[t] = vz;
  float mrun[8], lrun[8];
#pragma unroll
  for (int v = 0; v < 8; ++v) { mrun[v] = -3.0e38f; lrun[v] = 0.f; }

  stageKV(0, 0);
  wait_async0();
  __syncthreads();

#pragma unroll 1
  for (int kt0 = 0; kt0 < S; kt0 += 32) {
    const int buf = (kt0 >> 5) & 1;
    if (kt0 + 32 < S) stageKV(kt0 + 32, buf ^ 1);   // prefetch next block

    // ---- scores: Q(16x64) x K^T(64x32) -> two 16x16 tiles ----
    v8f sc0 = vz, sc1 = vz;
#pragma unroll
    for (int s2 = 0; s2 < 2; ++s2) {
      FragH kb0, kb1;
      const _Float16* p0 = &sK[buf][r        * KP + 32 * s2 + half * 8];
      const _Float16* p1 = &sK[buf][(16 + r) * KP + 32 * s2 + half * 8];
      kb0.q[0] = *(const uint4*)(p0); kb0.q[1] = *(const uint4*)(p0 + 16);
      kb1.q[0] = *(const uint4*)(p1); kb1.q[1] = *(const uint4*)(p1 + 16);
      sc0 = wmma16x16x32(qf[s2].v, kb0.v, sc0);
      sc1 = wmma16x16x32(qf[s2].v, kb1.v, sc1);
    }
    // additive key mask for this lane's two key columns
    const float mb0 = mk[kt0 + r]      ? 0.f : -1.0e30f;
    const float mb1 = mk[kt0 + 16 + r] ? 0.f : -1.0e30f;

    float e0a[8], e1a[8];
#pragma unroll
    for (int v = 0; v < 8; ++v) {
      e0a[v] = sc0[v] * 0.125f + mb0;   // 1/sqrt(HD)
      e1a[v] = sc1[v] * 0.125f + mb1;
    }
    // online softmax per row (row v+8*half lives within one 16-lane half)
#pragma unroll
    for (int v = 0; v < 8; ++v) {
      float mx = fmaxf(e0a[v], e1a[v]);
#pragma unroll
      for (int d2 = 1; d2 < 16; d2 <<= 1) mx = fmaxf(mx, __shfl_xor(mx, d2));
      const float mnew = fmaxf(mrun[v], mx);
      const float corr = __expf(mrun[v] - mnew);
      const float e0 = __expf(e0a[v] - mnew);
      const float e1 = __expf(e1a[v] - mnew);
      float ps = e0 + e1;
#pragma unroll
      for (int d2 = 1; d2 < 16; d2 <<= 1) ps += __shfl_xor(ps, d2);
      lrun[v] = lrun[v] * corr + ps;
      mrun[v] = mnew;
      e0a[v] = e0; e1a[v] = e1;
#pragma unroll
      for (int t = 0; t < 4; ++t) ctx[t][v] *= corr;
    }
    // relayout P (score C-layout) -> A-fragment layout via this wave's LDS
#pragma unroll
    for (int v = 0; v < 8; ++v) {
      _Float16* pr = &sP[wave][(v + 8 * half) * VP];
      pr[r]      = (_Float16)e0a[v];
      pr[16 + r] = (_Float16)e1a[v];
    }
    asm volatile("s_wait_dscnt 0x0" ::: "memory");  // same-wave DS RAW
    FragH pf;
    {
      const _Float16* p = &sP[wave][r * VP];
      pf.q[0] = *(const uint4*)(p + half * 8);
      pf.q[1] = *(const uint4*)(p + 16 + half * 8);
    }
    // ---- ctx += P(16x32) x V(32x64) ----
#pragma unroll
    for (int t = 0; t < 4; ++t) {
      FragH vb;
      const _Float16* p = &sV[buf][(16 * t + r) * VP + half * 8];
      vb.q[0] = *(const uint4*)(p); vb.q[1] = *(const uint4*)(p + 16);
      ctx[t] = wmma16x16x32(pf.v, vb.v, ctx[t]);
    }

    wait_async0();
    __syncthreads();
  }

  // normalize and scatter to [B*S, H] f16
  float invl[8];
#pragma unroll
  for (int v = 0; v < 8; ++v) invl[v] = 1.0f / lrun[v];
#pragma unroll
  for (int t = 0; t < 4; ++t)
#pragma unroll
    for (int v = 0; v < 8; ++v) {
      const int row = q0 + v + 8 * half;
      const int dd  = 16 * t + r;
      ctxh[((size_t)bb * S + row) * H + hh * HD + dd] =
          (_Float16)(ctx[t][v] * invl[v]);
    }
}

// ---------------------------------------------------------------------------
// LayerNorm over H=1024: one block per token, fp32 out + optional f16 copy.
// ---------------------------------------------------------------------------
__device__ __forceinline__ float block_sum(float v, float* sh) {
#pragma unroll
  for (int m = 1; m < 32; m <<= 1) v += __shfl_xor(v, m);
  const int w = threadIdx.x >> 5;
  if ((threadIdx.x & 31) == 0) sh[w] = v;
  __syncthreads();
  float t = 0.f;
#pragma unroll
  for (int i = 0; i < 8; ++i) t += sh[i];
  __syncthreads();
  return t;
}

__global__ __launch_bounds__(256)
void layernorm(const float* __restrict__ x, const float* __restrict__ g,
               const float* __restrict__ b, float* __restrict__ outF,
               _Float16* __restrict__ outH)
{
  __shared__ float sh[8];
  const int row = blockIdx.x;
  const float* xr = x + (size_t)row * H;
  float v[4];
  float s = 0.f;
#pragma unroll
  for (int i = 0; i < 4; ++i) { v[i] = xr[threadIdx.x + 256 * i]; s += v[i]; }
  s = block_sum(s, sh);
  const float mu = s * (1.0f / H);
  float sq = 0.f;
#pragma unroll
  for (int i = 0; i < 4; ++i) { const float d = v[i] - mu; sq += d * d; }
  sq = block_sum(sq, sh);
  const float inv = rsqrtf(sq * (1.0f / H) + LN_EPS);
#pragma unroll
  for (int i = 0; i < 4; ++i) {
    const int c = threadIdx.x + 256 * i;
    const float y = (v[i] - mu) * inv * g[c] + b[c];
    outF[(size_t)row * H + c] = y;
    if (outH) outH[(size_t)row * H + c] = (_Float16)y;
  }
}

// ---------------------------------------------------------------------------
// Converters
// ---------------------------------------------------------------------------
__global__ void f32_to_f16_kern(const float* __restrict__ in,
                                _Float16* __restrict__ out, int n) {
  const int i = blockIdx.x * blockDim.x + threadIdx.x;
  if (i < n) out[i] = (_Float16)in[i];
}

// in: [K,N] f32  ->  out: [N,K] f16  (tiled transpose)
__global__ __launch_bounds__(256)
void transpose_f32_to_f16(const float* __restrict__ in,
                          _Float16* __restrict__ out, int K, int N)
{
  __shared__ float tile[32][33];
  const int tx = threadIdx.x;
  const int ty = threadIdx.y;
  const int n0 = blockIdx.x * 32;
  const int k0 = blockIdx.y * 32;
#pragma unroll
  for (int i = 0; i < 4; ++i)
    tile[ty + 8 * i][tx] = in[(size_t)(k0 + ty + 8 * i) * N + n0 + tx];
  __syncthreads();
#pragma unroll
  for (int i = 0; i < 4; ++i)
    out[(size_t)(n0 + ty + 8 * i) * K + k0 + tx] =
        (_Float16)tile[tx][ty + 8 * i];
}

// ---------------------------------------------------------------------------
// Host launcher
// ---------------------------------------------------------------------------
extern "C" void kernel_launch(void* const* d_in, const int* in_sizes, int n_in,
                              void* d_out, int out_size, void* d_ws,
                              size_t ws_size, hipStream_t stream)
{
  (void)in_sizes; (void)n_in; (void)out_size; (void)ws_size;

  const float* x    = (const float*)d_in[0];
  const int*   msk  = (const int*)  d_in[1];
  const float* Wq   = (const float*)d_in[2];  const float* bq  = (const float*)d_in[3];
  const float* Wk   = (const float*)d_in[4];  const float* bk  = (const float*)d_in[5];
  const float* Wv   = (const float*)d_in[6];  const float* bv  = (const float*)d_in[7];
  const float* Wao  = (const float*)d_in[8];  const float* bao = (const float*)d_in[9];
  const float* g1   = (const float*)d_in[10]; const float* b1  = (const float*)d_in[11];
  const float* Wi   = (const float*)d_in[12]; const float* bi  = (const float*)d_in[13];
  const float* Wo   = (const float*)d_in[14]; const float* bo  = (const float*)d_in[15];
  const float* g2   = (const float*)d_in[16]; const float* b2  = (const float*)d_in[17];
  float* out = (float*)d_out;

  char* ws = (char*)d_ws;
  size_t off = 0;
  auto alloc = [&](size_t bytes) -> void* {
    off = (off + 255) & ~(size_t)255;
    void* p = ws + off;
    off += bytes;
    return p;
  };

  _Float16* xh      = (_Float16*)alloc((size_t)MT * H * 2);
  _Float16* WqT     = (_Float16*)alloc((size_t)H * H * 2);
  _Float16* WkT     = (_Float16*)alloc((size_t)H * H * 2);
  _Float16* WvT     = (_Float16*)alloc((size_t)H * H * 2);
  _Float16* WaoT    = (_Float16*)alloc((size_t)H * H * 2);
  _Float16* WiT     = (_Float16*)alloc((size_t)FF * H * 2);
  _Float16* WoT     = (_Float16*)alloc((size_t)H * FF * 2);
  _Float16* qhB     = (_Float16*)alloc((size_t)MT * H * 2);
  _Float16* khB     = (_Float16*)alloc((size_t)MT * H * 2);
  _Float16* vtB     = (_Float16*)alloc((size_t)MT * H * 2);
  _Float16* ctxh    = (_Float16*)alloc((size_t)MT * H * 2);
  float*    attnPre = (float*)   alloc((size_t)MT * H * 4);
  float*    attnOut = (float*)   alloc((size_t)MT * H * 4);
  _Float16* attnOutH= (_Float16*)alloc((size_t)MT * H * 2);
  _Float16* inter   = (_Float16*)alloc((size_t)MT * FF * 2);
  float*    ffnPre  = (float*)   alloc((size_t)MT * H * 4);

  // 1) convert activations + transpose-convert weights to f16
  f32_to_f16_kern<<<(MT * H + 255) / 256, 256, 0, stream>>>(x, xh, MT * H);
  dim3 tb(32, 8);
  transpose_f32_to_f16<<<dim3(H / 32, H / 32), tb, 0, stream>>>(Wq, WqT, H, H);
  transpose_f32_to_f16<<<dim3(H / 32, H / 32), tb, 0, stream>>>(Wk, WkT, H, H);
  transpose_f32_to_f16<<<dim3(H / 32, H / 32), tb, 0, stream>>>(Wv, WvT, H, H);
  transpose_f32_to_f16<<<dim3(H / 32, H / 32), tb, 0, stream>>>(Wao, WaoT, H, H);
  transpose_f32_to_f16<<<dim3(FF / 32, H / 32), tb, 0, stream>>>(Wi, WiT, H, FF);
  transpose_f32_to_f16<<<dim3(H / 32, FF / 32), tb, 0, stream>>>(Wo, WoT, FF, H);

  // 2) QKV projections (WMMA), scattering into attention layouts
  gemm_wmma<EPI_QK><<<dim3(H / TN, MT / TM), 256, 0, stream>>>(
      xh, WqT, bq, nullptr, nullptr, qhB, MT, H, H);
  gemm_wmma<EPI_QK><<<dim3(H / TN, MT / TM), 256, 0, stream>>>(
      xh, WkT, bk, nullptr, nullptr, khB, MT, H, H);
  gemm_wmma<EPI_VT><<<dim3(H / TN, MT / TM), 256, 0, stream>>>(
      xh, WvT, bv, nullptr, nullptr, vtB, MT, H, H);

  // 3) fused flash attention
  attn_flash<<<dim3(S / 128, Bb * NH), 256, 0, stream>>>(qhB, khB, vtB, msk, ctxh);

  // 4) output projection + residual, LN1
  gemm_wmma<EPI_RES><<<dim3(H / TN, MT / TM), 256, 0, stream>>>(
      ctxh, WaoT, bao, x, attnPre, nullptr, MT, H, H);
  layernorm<<<MT, 256, 0, stream>>>(attnPre, g1, b1, attnOut, attnOutH);

  // 5) FFN: GELU(attn_out @ Wi + bi) @ Wo + bo + attn_out, LN2 -> d_out
  gemm_wmma<EPI_GELU><<<dim3(FF / TN, MT / TM), 256, 0, stream>>>(
      attnOutH, WiT, bi, nullptr, nullptr, inter, MT, FF, H);
  gemm_wmma<EPI_RES><<<dim3(H / TN, MT / TM), 256, 0, stream>>>(
      inter, WoT, bo, attnOut, ffnPre, nullptr, MT, H, FF);
  layernorm<<<MT, 256, 0, stream>>>(ffnPre, g2, b2, out, nullptr);
}
